// RFP_UAGs_50929722196517
// MI455X (gfx1250) — compile-verified, matrix operand order checked by reading, not run on hardware
//
#include <hip/hip_runtime.h>
#include <hip/hip_bf16.h>

typedef __attribute__((ext_vector_type(16))) __bf16 v16bf;
typedef __attribute__((ext_vector_type(8)))  __bf16 v8bf;
typedef __attribute__((ext_vector_type(2)))  __bf16 v2bf;
typedef __attribute__((ext_vector_type(8)))  float  v8f;

namespace {
constexpr int Cc = 128, Hh = 96, Ww = 96, Dd = 4;
constexpr int CS  = Hh * Ww * Dd;   // channel stride in x/out   = 36864
constexpr int BS  = Cc * CS;        // batch stride              = 4718592
constexpr int HS  = Ww * Dd;        // row (h) stride            = 384
constexpr int STR = 136;            // padded LDS row stride (bf16) -> 272B
}

// Load a 16-element bf16 WMMA fragment from LDS: two contiguous 16B chunks
// (K = base..base+7 and K = base+16..base+23 per the CDNA5 16-bit A/B layout).
__device__ __forceinline__ v16bf lds_frag(const __bf16* p) {
  v8bf a = *(const v8bf*)(p);
  v8bf b = *(const v8bf*)(p + 16);
  return __builtin_shufflevector(a, b, 0,1,2,3,4,5,6,7,8,9,10,11,12,13,14,15);
}

// Build a B-matrix fragment (activations) straight from global f32 data:
// lane (l,hf) element e -> channel c = 32*kt + 8*hf + e (chunk0), +16 (chunk1).
__device__ __forceinline__ v16bf gbl_frag(const float* __restrict__ p, int kt, int hf) {
  v16bf f;
  const int c0 = kt * 32 + hf * 8;
#pragma unroll
  for (int e = 0; e < 8; ++e) f[e]     = (__bf16)p[(c0 + e) * CS];
#pragma unroll
  for (int e = 0; e < 8; ++e) f[8 + e] = (__bf16)p[(c0 + 16 + e) * CS];
  return f;
}

// One recurrence-step GEMM pair: acc += sA @ curF + sB @ prevF, with the
// A-fragment loads batched 8 at a time so ds_load latency pipelines across
// the WMMA chain instead of serializing in front of each one.
__device__ __forceinline__ void step_gemms(const __bf16* sA, const __bf16* sB,
                                           const v16bf* curF, const v16bf* prevF,
                                           v8f* acc, int l, int hf) {
#pragma unroll
  for (int kt = 0; kt < 4; ++kt) {
    const int koff = kt * 32 + hf * 8;
    v16bf a[8];
#pragma unroll
    for (int mt = 0; mt < 8; ++mt) a[mt] = lds_frag(&sA[(mt * 16 + l) * STR + koff]);
#pragma unroll
    for (int mt = 0; mt < 8; ++mt)
      acc[mt] = __builtin_amdgcn_wmma_f32_16x16x32_bf16(
          false, a[mt], false, curF[kt], (short)0, acc[mt], false, false);
#pragma unroll
    for (int mt = 0; mt < 8; ++mt) a[mt] = lds_frag(&sB[(mt * 16 + l) * STR + koff]);
#pragma unroll
    for (int mt = 0; mt < 8; ++mt)
      acc[mt] = __builtin_amdgcn_wmma_f32_16x16x32_bf16(
          false, a[mt], false, prevF[kt], (short)0, acc[mt], false, false);
  }
}

// ---------------------------------------------------------------------------
// Phase A: row scan.  hs[:, :, h, :] = relu(W1 @ x_row_h + W2 @ prev + b1+b2)
// hs row 0 = x row 0 (no relu).  hs is written into `out` (same layout as x).
// ---------------------------------------------------------------------------
__global__ __launch_bounds__(128)
void rowscan_kernel(const float* __restrict__ x,
                    const float* __restrict__ W1, const float* __restrict__ b1,
                    const float* __restrict__ W2, const float* __restrict__ b2,
                    float* __restrict__ out) {
  __shared__ __bf16 sA[Cc * STR];      // W1 (bf16, row-major o x c, padded)
  __shared__ __bf16 sB[Cc * STR];      // W2
  __shared__ float  sbias[Cc];         // b1 + b2
  __shared__ __bf16 sStg[4][16 * STR]; // per-wave prev staging (16 cols x 128 ch)

  const int d    = blockIdx.y;
  const int tile = blockIdx.x;                  // 0..11
  const int t    = threadIdx.x;
  const int wave = t >> 5, lane = t & 31;
  const int l = lane & 15, hf = lane >> 4;

  // Cooperative weight load + f32->bf16 convert (c = t, coalesced).
  const float* W1d = W1 + d * Cc * Cc;
  const float* W2d = W2 + d * Cc * Cc;
  for (int o = 0; o < Cc; ++o) {
    sA[o * STR + t] = (__bf16)W1d[o * Cc + t];
    sB[o * STR + t] = (__bf16)W2d[o * Cc + t];
  }
  sbias[t] = b1[d * Cc + t] + b2[d * Cc + t];
  __syncthreads();

  // This lane's column: (b, w) pair.  16 consecutive w per wave -> coalesced.
  const int col  = tile * 64 + wave * 16 + l;   // 0..767
  const int bb   = col / Ww, wcol = col % Ww;
  const int base = bb * BS + wcol * Dd + d;     // + c*CS + h*HS

  __bf16* stg = &sStg[wave][0];

  // prev = row 0 of x (no relu); also copy row 0 into out (hs row 0).
  v16bf prevF[4];
  {
    const float* p0 = x + base;
#pragma unroll
    for (int kt = 0; kt < 4; ++kt) prevF[kt] = gbl_frag(p0, kt, hf);
#pragma unroll
    for (int i = 0; i < 64; ++i) {
      const int c = hf * 64 + i;
      out[base + c * CS] = x[base + c * CS];
    }
  }

  for (int h = 1; h < Hh; ++h) {
    const float* px = x + base + h * HS;

    // Prefetch next row's x (global_prefetch_b8).
    if (h + 1 < Hh) {
#pragma unroll
      for (int kt = 0; kt < 4; ++kt)
        __builtin_prefetch(px + HS + (kt * 32 + hf * 8) * CS, 0, 1);
    }

    v16bf rowF[4];
#pragma unroll
    for (int kt = 0; kt < 4; ++kt) rowF[kt] = gbl_frag(px, kt, hf);

    // acc init = bias (C/D layout: element r -> channel 16*mt + 8*hf + r).
    v8f acc[8];
#pragma unroll
    for (int mt = 0; mt < 8; ++mt)
#pragma unroll
      for (int r = 0; r < 8; ++r) acc[mt][r] = sbias[mt * 16 + hf * 8 + r];

    step_gemms(sA, sB, rowF, prevF, acc, l, hf);

    // relu -> write hs row to out (f32) + stage bf16 for next step's prev.
    float* po = out + base + h * HS;
#pragma unroll
    for (int mt = 0; mt < 8; ++mt) {
#pragma unroll
      for (int r = 0; r < 8; r += 2) {
        const int c = mt * 16 + hf * 8 + r;
        const float v0 = fmaxf(acc[mt][r], 0.f);
        const float v1 = fmaxf(acc[mt][r + 1], 0.f);
        po[c * CS]       = v0;
        po[(c + 1) * CS] = v1;
        v2bf pk; pk.x = (__bf16)v0; pk.y = (__bf16)v1;
        *(v2bf*)&stg[l * STR + c] = pk;
      }
    }
    __syncthreads();
#pragma unroll
    for (int kt = 0; kt < 4; ++kt)
      prevF[kt] = lds_frag(&stg[l * STR + kt * 32 + hf * 8]);
  }
}

// ---------------------------------------------------------------------------
// Phase B: column scan over hs (stored in `out`), producing out = x + hse.
// col0: out = x + relu(hs_col0), prev = relu(hs_col0).
// w>=1: new = relu(W4 @ hs_col_w + W5 @ prev + b4+b5); out = x + new.
// ---------------------------------------------------------------------------
__global__ __launch_bounds__(128)
void colscan_kernel(const float* __restrict__ x,
                    const float* __restrict__ W4, const float* __restrict__ b4,
                    const float* __restrict__ W5, const float* __restrict__ b5,
                    float* __restrict__ out) {
  __shared__ __bf16 sA[Cc * STR];
  __shared__ __bf16 sB[Cc * STR];
  __shared__ float  sbias[Cc];
  __shared__ __bf16 sStg[4][16 * STR];

  const int d    = blockIdx.y;
  const int tile = blockIdx.x;
  const int t    = threadIdx.x;
  const int wave = t >> 5, lane = t & 31;
  const int l = lane & 15, hf = lane >> 4;

  const float* W4d = W4 + d * Cc * Cc;
  const float* W5d = W5 + d * Cc * Cc;
  for (int o = 0; o < Cc; ++o) {
    sA[o * STR + t] = (__bf16)W4d[o * Cc + t];
    sB[o * STR + t] = (__bf16)W5d[o * Cc + t];
  }
  sbias[t] = b4[d * Cc + t] + b5[d * Cc + t];
  __syncthreads();

  // This lane's column: (b, h) pair.
  const int col  = tile * 64 + wave * 16 + l;   // 0..767
  const int bb   = col / Hh, hcol = col % Hh;
  const int base = bb * BS + hcol * HS + d;     // + c*CS + w*Dd

  __bf16* stg = &sStg[wave][0];

  // col 0: prev = relu(hs), out = x + relu(hs).  Each (lane,half) owns the
  // channel set it reads (c%16 in [8*hf, 8*hf+8)), so read-then-write is safe.
  v16bf prevF[4];
  {
    float* ph = out + base;           // hs column w = 0
#pragma unroll
    for (int kt = 0; kt < 4; ++kt) {
      v16bf f;
      const int c0 = kt * 32 + hf * 8;
#pragma unroll
      for (int e = 0; e < 8; ++e) {
        const int i0 = (c0 + e) * CS;
        const float r0 = fmaxf(ph[i0], 0.f);
        f[e] = (__bf16)r0;
        ph[i0] = x[base + i0] + r0;
        const int i1 = (c0 + 16 + e) * CS;
        const float r1 = fmaxf(ph[i1], 0.f);
        f[8 + e] = (__bf16)r1;
        ph[i1] = x[base + i1] + r1;
      }
      prevF[kt] = f;
    }
  }

  for (int w = 1; w < Ww; ++w) {
    const float* ph = out + base + w * Dd;      // hs column w (raw, no relu)

    if (w + 1 < Ww) {
#pragma unroll
      for (int kt = 0; kt < 4; ++kt)
        __builtin_prefetch(ph + Dd + (kt * 32 + hf * 8) * CS, 0, 1);
    }

    v16bf hsF[4];
#pragma unroll
    for (int kt = 0; kt < 4; ++kt) hsF[kt] = gbl_frag(ph, kt, hf);

    v8f acc[8];
#pragma unroll
    for (int mt = 0; mt < 8; ++mt)
#pragma unroll
      for (int r = 0; r < 8; ++r) acc[mt][r] = sbias[mt * 16 + hf * 8 + r];

    step_gemms(sA, sB, hsF, prevF, acc, l, hf);

    float* po = out + base + w * Dd;
#pragma unroll
    for (int mt = 0; mt < 8; ++mt) {
#pragma unroll
      for (int r = 0; r < 8; r += 2) {
        const int c = mt * 16 + hf * 8 + r;
        const float v0 = fmaxf(acc[mt][r], 0.f);
        const float v1 = fmaxf(acc[mt][r + 1], 0.f);
        po[c * CS]       = x[base + w * Dd + c * CS] + v0;
        po[(c + 1) * CS] = x[base + w * Dd + (c + 1) * CS] + v1;
        v2bf pk; pk.x = (__bf16)v0; pk.y = (__bf16)v1;
        *(v2bf*)&stg[l * STR + c] = pk;
      }
    }
    __syncthreads();
#pragma unroll
    for (int kt = 0; kt < 4; ++kt)
      prevF[kt] = lds_frag(&stg[l * STR + kt * 32 + hf * 8]);
  }
}

extern "C" void kernel_launch(void* const* d_in, const int* in_sizes, int n_in,
                              void* d_out, int out_size, void* d_ws, size_t ws_size,
                              hipStream_t stream) {
  const float* x  = (const float*)d_in[0];
  const float* W1 = (const float*)d_in[1];
  const float* b1 = (const float*)d_in[2];
  const float* W2 = (const float*)d_in[3];
  const float* b2 = (const float*)d_in[4];
  const float* W4 = (const float*)d_in[5];
  const float* b4 = (const float*)d_in[6];
  const float* W5 = (const float*)d_in[7];
  const float* b5 = (const float*)d_in[8];
  float* out = (float*)d_out;

  dim3 grid(12, Dd), block(128);
  rowscan_kernel<<<grid, block, 0, stream>>>(x, W1, b1, W2, b2, out);
  colscan_kernel<<<grid, block, 0, stream>>>(x, W4, b4, W5, b5, out);
}
